// AutoRegressiveLSTM_77661598646770
// MI455X (gfx1250) — matrix operand order repaired and checked by reading, not compiled, not run on hardware
//
#include <hip/hip_runtime.h>

// Problem dimensions (fixed by the reference).
#define T_STEPS 512
#define BATCH   512
#define IN_DIM  128
#define HID     1024
#define OUT_DIM 128
#define KCAT    (IN_DIM + OUT_DIM + HID)   // 1280: [x | z | h]
#define GATES   (4 * HID)                  // 4096
#define KSTEPS  (KCAT / 32)                // 40

typedef __attribute__((ext_vector_type(16))) __bf16 v16bf;
typedef __attribute__((ext_vector_type(8)))  __bf16 v8bf;
typedef __attribute__((ext_vector_type(8)))  float  v8f;
typedef __attribute__((ext_vector_type(4)))  float  v4f;
typedef __attribute__((ext_vector_type(4)))  int    v4i;

#if __has_builtin(__builtin_amdgcn_global_load_async_to_lds_b128)
#define HAVE_ASYNC_LDS 1
typedef __attribute__((address_space(1))) v4i* as1_v4i_ptr;  // global (prints as __device__)
typedef __attribute__((address_space(3))) v4i* as3_v4i_ptr;  // LDS
#else
#define HAVE_ASYNC_LDS 0
#endif

// Copy 16 bytes global -> LDS. Async (ASYNCcnt-tracked) when available.
static __device__ __forceinline__ void copy16_g2lds(const __bf16* g, __bf16* l) {
#if HAVE_ASYNC_LDS
    __builtin_amdgcn_global_load_async_to_lds_b128((as1_v4i_ptr)g, (as3_v4i_ptr)l, 0, 0);
#else
    *(v8bf*)l = *(const v8bf*)g;   // sync fallback: global_load + ds_store
#endif
}

static __device__ __forceinline__ void async_wait_le2() {
#if HAVE_ASYNC_LDS
#if __has_builtin(__builtin_amdgcn_s_wait_asynccnt)
    __builtin_amdgcn_s_wait_asynccnt(2);
#else
    asm volatile("s_wait_asynccnt 0x2" ::: "memory");
#endif
#endif
}
static __device__ __forceinline__ void async_wait_le0() {
#if HAVE_ASYNC_LDS
#if __has_builtin(__builtin_amdgcn_s_wait_asynccnt)
    __builtin_amdgcn_s_wait_asynccnt(0);
#else
    asm volatile("s_wait_asynccnt 0x0" ::: "memory");
#endif
#endif
}

static __device__ __forceinline__ v16bf concat8(v8bf lo, v8bf hi) {
    return __builtin_shufflevector(lo, hi, 0,1,2,3,4,5,6,7,8,9,10,11,12,13,14,15);
}

static __device__ __forceinline__ v8bf load8_bf(const __bf16* p) {
    return *(const v8bf*)p;
}

// Load 8 contiguous f32, RNE-convert to bf16 in-register.
static __device__ __forceinline__ v8bf load8_f32_as_bf(const float* p) {
    v4f a = *(const v4f*)p;
    v4f b = *(const v4f*)(p + 4);
    v8bf r;
    r[0] = (__bf16)a[0]; r[1] = (__bf16)a[1]; r[2] = (__bf16)a[2]; r[3] = (__bf16)a[3];
    r[4] = (__bf16)b[0]; r[5] = (__bf16)b[1]; r[6] = (__bf16)b[2]; r[7] = (__bf16)b[3];
    return r;
}

// A-matrix 16x32 bf16 fragment (ISA 7.12.2): lane L holds row M=L%16.
// kbase already includes the (lane&16 ? 8 : 0) shift; elems 0..7 at kbase,
// elems 8..15 at kbase+16.
static __device__ __forceinline__ v16bf afrag_bf(const __bf16* row, int kbase) {
    return concat8(load8_bf(row + kbase), load8_bf(row + kbase + 16));
}
static __device__ __forceinline__ v16bf afrag_f32(const float* row, int kbase) {
    return concat8(load8_f32_as_bf(row + kbase), load8_f32_as_bf(row + kbase + 16));
}

// B-matrix 32x16 bf16 fragment: lane L holds column N=L%16; kbase already
// includes the (lane&16 ? 8 : 0) shift; elems 0..7 at kbase, 8..15 at kbase+16.
static __device__ __forceinline__ v16bf bfrag_bf(const __bf16* col, int kbase) {
    return concat8(load8_bf(col + kbase), load8_bf(col + kbase + 16));
}

static __device__ __forceinline__ v8f wmma_bf16(v16bf a, v16bf b, v8f c) {
    return __builtin_amdgcn_wmma_f32_16x16x32_bf16(false, a, false, b, (short)0, c,
                                                   false, false);
}

static __device__ __forceinline__ float sigmoidf_(float x) {
    return 1.0f / (1.0f + __expf(-x));
}

// ---------------------------------------------------------------------------
// Kernel A: fused gates GEMM [512 x (1024x4gates), K=1280] + LSTM cell.
// Block = 256 threads (8 waves, 4m x 2n). Wave tile: M=32 (two 16-row
// subtiles) x N=16 hidden cols x 4 gates -> 8 accumulators/wave.
// Block tile: M=128, N=32. Grid: (BATCH/128, HID/32) = (4, 32).
// B (weight) tile for the current k-step (128 rows x 32 k = 8 KB) is staged
// in LDS, double-buffered, via async global->LDS copies (ASYNCcnt pipeline).
// ---------------------------------------------------------------------------
__global__ __launch_bounds__(256)
void arlstm_gates_step(const float*  __restrict__ x_t,   // [B, IN_DIM]
                       const __bf16* __restrict__ z_bf,  // [B, OUT_DIM] prev z
                       const __bf16* __restrict__ h_in,  // [B, HID]     prev h
                       const __bf16* __restrict__ Wc,    // [4H, KCAT]
                       const float*  __restrict__ b_ih,  // [4H]
                       const float*  __restrict__ b_hh,  // [4H]
                       float*        __restrict__ c_st,  // [B, HID] (in/out)
                       __bf16*       __restrict__ h_out, // [B, HID] new h bf16
                       float*        __restrict__ h_f32) // [B, HID] new h f32
{
    // LDS: double-buffered B tile. Row r (= gate*32 + col-in-block) holds the
    // 32 contiguous K values (64 B) of weight row for that output column.
    __shared__ __bf16 lds_b[2][128][32];   // 16 KB

    const int lane   = threadIdx.x & 31;
    const int wave   = threadIdx.x >> 5;
    const int wm     = wave >> 1;          // 0..3
    const int wn     = wave & 1;           // 0..1
    const int m0     = blockIdx.x * 128 + wm * 32;
    const int ny0    = blockIdx.y * 32;    // block's first hidden column
    const int nl     = lane & 15;
    const int hihalf = (lane >> 4) & 1;
    const int abase  = hihalf ? 8 : 0;     // A-fragment K sub-offset
    const int bbase  = hihalf ? 8 : 0;     // B-fragment K sub-offset

    // A-row pointers for the two 16-row subtiles.
    const int ar0 = m0 + nl;
    const int ar1 = m0 + 16 + nl;
    const float*  xr0 = x_t  + (size_t)ar0 * IN_DIM;
    const float*  xr1 = x_t  + (size_t)ar1 * IN_DIM;
    const __bf16* zr0 = z_bf + (size_t)ar0 * OUT_DIM;
    const __bf16* zr1 = z_bf + (size_t)ar1 * OUT_DIM;
    const __bf16* hr0 = h_in + (size_t)ar0 * HID;
    const __bf16* hr1 = h_in + (size_t)ar1 * HID;

    // Stage the 8 KB B tile for k-step s into lds_b[s&1]:
    // 512 chunks of 16 B; each of the 256 threads moves 2 chunks.
    auto stage_step = [&](int s) {
        if (s < KSTEPS) {
            const int k0  = s * 32;
            const int buf = s & 1;
            #pragma unroll
            for (int half = 0; half < 2; ++half) {
                const int ch    = threadIdx.x + half * 256;   // 0..511
                const int lrow  = ch >> 2;                    // 0..127
                const int piece = ch & 3;                     // 16B piece in row
                const int wrow  = (lrow >> 5) * HID + ny0 + (lrow & 31);
                const __bf16* g = Wc + (size_t)wrow * KCAT + k0 + piece * 8;
                copy16_g2lds(g, &lds_b[buf][lrow][piece * 8]);
            }
        }
    };

    v8f acc[2][4];
    #pragma unroll
    for (int s = 0; s < 2; ++s)
        #pragma unroll
        for (int g = 0; g < 4; ++g)
            acc[s][g] = (v8f){0,0,0,0,0,0,0,0};

    stage_step(0);
    stage_step(1);

    for (int s = 0; s < KSTEPS; ++s) {
        const int k0  = s * 32;
        const int buf = s & 1;

        if (s < KSTEPS - 1) async_wait_le2();   // copies for step s complete
        else                async_wait_le0();
        __syncthreads();                        // ... in every wave

        // A fragments (direct from global; region depends on k0).
        v16bf a0, a1;
        if (k0 < IN_DIM) {
            a0 = afrag_f32(xr0, k0 + abase);
            a1 = afrag_f32(xr1, k0 + abase);
        } else if (k0 < IN_DIM + OUT_DIM) {
            a0 = afrag_bf(zr0, (k0 - IN_DIM) + abase);
            a1 = afrag_bf(zr1, (k0 - IN_DIM) + abase);
        } else {
            a0 = afrag_bf(hr0, (k0 - (IN_DIM + OUT_DIM)) + abase);
            a1 = afrag_bf(hr1, (k0 - (IN_DIM + OUT_DIM)) + abase);
        }

        // B fragments from LDS; 2 WMMAs (two M-subtiles) per gate.
        #pragma unroll
        for (int g = 0; g < 4; ++g) {
            const __bf16* lb = &lds_b[buf][g * 32 + wn * 16 + nl][0];
            v16bf bfr = bfrag_bf(lb, bbase);
            acc[0][g] = wmma_bf16(a0, bfr, acc[0][g]);
            acc[1][g] = wmma_bf16(a1, bfr, acc[1][g]);
        }

        __syncthreads();          // all waves done reading lds_b[buf]
        stage_step(s + 2);        // refill the freed buffer
    }

    // LSTM cell elementwise, in-register (torch gate order i,f,g,o).
    const int nc = ny0 + wn * 16 + nl;
    const float bi = b_ih[nc]           + b_hh[nc];
    const float bf = b_ih[nc + 1 * HID] + b_hh[nc + 1 * HID];
    const float bg = b_ih[nc + 2 * HID] + b_hh[nc + 2 * HID];
    const float bo = b_ih[nc + 3 * HID] + b_hh[nc + 3 * HID];
    const int mshift = hihalf ? 8 : 0;   // C/D layout: lanes 16-31 hold M=8..15

    #pragma unroll
    for (int sub = 0; sub < 2; ++sub) {
        #pragma unroll
        for (int r = 0; r < 8; ++r) {
            const int m = m0 + sub * 16 + mshift + r;
            const size_t idx = (size_t)m * HID + nc;
            const float gi = sigmoidf_(acc[sub][0][r] + bi);
            const float gf = sigmoidf_(acc[sub][1][r] + bf);
            const float gg = tanhf(acc[sub][2][r] + bg);
            const float go = sigmoidf_(acc[sub][3][r] + bo);
            const float cn = gf * c_st[idx] + gi * gg;
            const float hn = go * tanhf(cn);
            c_st[idx]  = cn;
            h_f32[idx] = hn;
            h_out[idx] = (__bf16)hn;
        }
    }
}

// ---------------------------------------------------------------------------
// Kernel B: fc projection z = tanh(h @ W_fc^T + b_fc) * (t < seq_len).
// M=512, N=128, K=1024. 8 waves x (16 rows x 16 cols). Grid (4, 8).
// ---------------------------------------------------------------------------
__global__ __launch_bounds__(256)
void arlstm_fc_step(const __bf16* __restrict__ h_in,   // [B, HID]
                    const __bf16* __restrict__ Wfc,    // [O, HID]
                    const float*  __restrict__ b_fc,   // [O]
                    const int*    __restrict__ seqlen, // [B]
                    int t,
                    float*        __restrict__ out_t,  // d_out + t*B*O
                    __bf16*       __restrict__ z_bf)   // [B, O] feedback
{
    const int lane   = threadIdx.x & 31;
    const int wave   = threadIdx.x >> 5;
    const int m0     = blockIdx.x * 128 + wave * 16;
    const int n0     = blockIdx.y * 16;
    const int nl     = lane & 15;
    const int hihalf = (lane >> 4) & 1;
    const int abase  = hihalf ? 8 : 0;
    const int bbase  = hihalf ? 8 : 0;

    const __bf16* hr = h_in + (size_t)(m0 + nl) * HID;
    const __bf16* wr = Wfc  + (size_t)(n0 + nl) * HID;

    v8f acc = {0,0,0,0,0,0,0,0};
    for (int k0 = 0; k0 < HID; k0 += 32) {
        acc = wmma_bf16(afrag_bf(hr, k0 + abase), bfrag_bf(wr, k0 + bbase), acc);
    }

    const float bn = b_fc[n0 + nl];
    const int mshift = hihalf ? 8 : 0;
    #pragma unroll
    for (int r = 0; r < 8; ++r) {
        const int m = m0 + mshift + r;
        const float active = (t < seqlen[m]) ? 1.0f : 0.0f;
        const float zv = tanhf(acc[r] + bn) * active;
        const size_t idx = (size_t)m * OUT_DIM + (n0 + nl);
        out_t[idx] = zv;
        z_bf[idx]  = (__bf16)zv;
    }
}

// ---------------------------------------------------------------------------
// One-time init: combined bf16 weights Wc = [W_ih | W_hh], bf16 W_fc.
// ---------------------------------------------------------------------------
__global__ void arlstm_init_weights(const float* __restrict__ W_ih,  // [4H, 256]
                                    const float* __restrict__ W_hh,  // [4H, 1024]
                                    const float* __restrict__ W_fc,  // [O, 1024]
                                    __bf16* __restrict__ Wc,         // [4H, 1280]
                                    __bf16* __restrict__ Wfc)        // [O, 1024]
{
    const int stride = gridDim.x * blockDim.x;
    const int total1 = GATES * KCAT;
    for (int i = blockIdx.x * blockDim.x + threadIdx.x; i < total1; i += stride) {
        const int r = i / KCAT;
        const int k = i - r * KCAT;
        const float v = (k < IN_DIM + OUT_DIM)
                          ? W_ih[(size_t)r * (IN_DIM + OUT_DIM) + k]
                          : W_hh[(size_t)r * HID + (k - (IN_DIM + OUT_DIM))];
        Wc[i] = (__bf16)v;
    }
    const int total2 = OUT_DIM * HID;
    for (int i = blockIdx.x * blockDim.x + threadIdx.x; i < total2; i += stride) {
        Wfc[i] = (__bf16)W_fc[i];
    }
}

__global__ void arlstm_init_state(const float* __restrict__ h0,
                                  const float* __restrict__ c0,
                                  const float* __restrict__ z0,
                                  __bf16* __restrict__ h_bf,
                                  float*  __restrict__ h_f32,
                                  float*  __restrict__ c_ws,
                                  __bf16* __restrict__ z_bf)
{
    const int stride = gridDim.x * blockDim.x;
    for (int i = blockIdx.x * blockDim.x + threadIdx.x; i < BATCH * HID; i += stride) {
        h_bf[i]  = (__bf16)h0[i];
        h_f32[i] = h0[i];
        c_ws[i]  = c0[i];
    }
    for (int i = blockIdx.x * blockDim.x + threadIdx.x; i < BATCH * OUT_DIM; i += stride) {
        z_bf[i] = (__bf16)z0[i];
    }
}

__global__ void arlstm_finalize(const float* __restrict__ h_f32,
                                const float* __restrict__ c_ws,
                                float* __restrict__ out_tail)  // d_out + T*B*O
{
    const int stride = gridDim.x * blockDim.x;
    for (int i = blockIdx.x * blockDim.x + threadIdx.x; i < BATCH * HID; i += stride) {
        out_tail[i]               = h_f32[i];
        out_tail[BATCH * HID + i] = c_ws[i];
    }
}

// ---------------------------------------------------------------------------
// Host driver: 2 kernels per time step on `stream` (graph-capture safe).
// ---------------------------------------------------------------------------
extern "C" void kernel_launch(void* const* d_in, const int* in_sizes, int n_in,
                              void* d_out, int out_size, void* d_ws, size_t ws_size,
                              hipStream_t stream) {
    const float* inputs = (const float*)d_in[0];   // [T, B, I]
    const float* h0     = (const float*)d_in[1];   // [B, H]
    const float* c0     = (const float*)d_in[2];   // [B, H]
    const float* z0     = (const float*)d_in[3];   // [B, O]
    const int*   seqlen = (const int*)  d_in[4];   // [B]
    const float* W_ih   = (const float*)d_in[5];   // [4H, I+O]
    const float* W_hh   = (const float*)d_in[6];   // [4H, H]
    const float* b_ih   = (const float*)d_in[7];   // [4H]
    const float* b_hh   = (const float*)d_in[8];   // [4H]
    const float* W_fc   = (const float*)d_in[9];   // [O, H]
    const float* b_fc   = (const float*)d_in[10];  // [O]

    char* p = (char*)d_ws;
    auto carve = [&p](size_t bytes) {
        void* q = (void*)p;
        p += (bytes + 255) & ~(size_t)255;
        return q;
    };
    __bf16* Wc    = (__bf16*)carve((size_t)GATES * KCAT * sizeof(__bf16));
    __bf16* Wfc   = (__bf16*)carve((size_t)OUT_DIM * HID * sizeof(__bf16));
    __bf16* h_bf0 = (__bf16*)carve((size_t)BATCH * HID * sizeof(__bf16));
    __bf16* h_bf1 = (__bf16*)carve((size_t)BATCH * HID * sizeof(__bf16));
    __bf16* z_bf  = (__bf16*)carve((size_t)BATCH * OUT_DIM * sizeof(__bf16));
    float*  c_ws  = (float*) carve((size_t)BATCH * HID * sizeof(float));
    float*  h_f32 = (float*) carve((size_t)BATCH * HID * sizeof(float));

    arlstm_init_weights<<<2048, 256, 0, stream>>>(W_ih, W_hh, W_fc, Wc, Wfc);
    arlstm_init_state<<<1024, 256, 0, stream>>>(h0, c0, z0, h_bf0, h_f32, c_ws, z_bf);

    float* out = (float*)d_out;
    const dim3 gridA(BATCH / 128, HID / 32);     // 4 x 32 workgroups
    const dim3 gridB(BATCH / 128, OUT_DIM / 16); // 4 x 8 workgroups

    for (int t = 0; t < T_STEPS; ++t) {
        const __bf16* h_in  = (t & 1) ? h_bf1 : h_bf0;
        __bf16*       h_out = (t & 1) ? h_bf0 : h_bf1;
        arlstm_gates_step<<<gridA, 256, 0, stream>>>(
            inputs + (size_t)t * BATCH * IN_DIM, z_bf, h_in, Wc, b_ih, b_hh,
            c_ws, h_out, h_f32);
        arlstm_fc_step<<<gridB, 256, 0, stream>>>(
            h_out, Wfc, b_fc, seqlen, t,
            out + (size_t)t * BATCH * OUT_DIM, z_bf);
    }

    arlstm_finalize<<<512, 256, 0, stream>>>(h_f32, c_ws,
                                             out + (size_t)T_STEPS * BATCH * OUT_DIM);
}
